// MoCo_62577673503718
// MI455X (gfx1250) — compile-verified, matrix operand order checked by reading, not run on hardware
//
#include <hip/hip_runtime.h>
#include <hip/hip_bf16.h>
#include <math.h>

// ---------------------------------------------------------------------------
// Problem constants (from reference)
// ---------------------------------------------------------------------------
#define NNODE   20000
#define NFEAT   128
#define NEDGE   320000
#define NEMB    128
#define NOUT    64
#define BATCH   512
#define QSIZE   2048
#define NQUEUE  16
#define MMT     0.999f
#define INV_TEMP (1.0f/0.07f)
#define NEG_SLOPE 0.2f

typedef __attribute__((ext_vector_type(16))) __bf16 v16bf;
typedef __attribute__((ext_vector_type(8)))  float  v8f;

// ---------------------------------------------------------------------------
// Weight pack: fp32 row-major W[K,M] -> bf16 WMMA B-fragments.
// Fragment (colTile, kt) holds 32 lanes x 16 bf16 contiguous (32B per lane):
//   lane: n = colTile*16 + (lane&15), hi = lane>>4
//   elem e (v=e/2, p=e%2): k = kt*32 + ((v<4)? 2v : 16+2(v-4)) + hi*8 + p
// ---------------------------------------------------------------------------
__global__ void pack_w_kernel(const float* __restrict__ W, __bf16* __restrict__ Wp,
                              int K, int M) {
  const int ntiles = (M / 16) * (K / 32);
  int i = blockIdx.x * blockDim.x + threadIdx.x;       // one thread per (tile, lane)
  if (i >= ntiles * 32) return;
  const int lane = i & 31;
  const int tile = i >> 5;
  const int KT = K / 32;
  const int kt = tile % KT;
  const int colTile = tile / KT;
  const int l16 = lane & 15, hi = lane >> 4;
  const int col = colTile * 16 + l16;
  __bf16* dst = Wp + (size_t)i * 16;
#pragma unroll
  for (int v = 0; v < 8; ++v) {
    const int kb = ((v < 4) ? (2 * v) : (16 + 2 * (v - 4))) + hi * 8;
    const int k = kt * 32 + kb;
    dst[2 * v]     = (__bf16)W[(size_t)k * M + col];
    dst[2 * v + 1] = (__bf16)W[(size_t)(k + 1) * M + col];
  }
}

// ---------------------------------------------------------------------------
// WMMA GEMM: C[N,M] = A[N,K] @ W[K,M] (+bias)(+relu), bf16 MACs / f32 accum.
// Block = 8 waves = 128 rows. Each wave computes a 16x64 strip: one A fragment
// (4x float4 contiguous loads) reused across 4 packed-B fragments (contiguous
// 32B loads) -> 4 v_wmma per K-step. act: 0=none, 2=relu.
// ---------------------------------------------------------------------------
__global__ void gemm_wmma_bf16(const float* __restrict__ A,
                               const __bf16* __restrict__ Wp,
                               const float* __restrict__ bias,
                               float* __restrict__ C,
                               int Nrow, int K, int M, int act) {
  const int wave = threadIdx.x >> 5;
  const int lane = threadIdx.x & 31;
  const int l16  = lane & 15;
  const int hi   = lane >> 4;
  const int colBase = blockIdx.x * 64;
  const int ct0 = colBase >> 4;
  const int rowBase = (blockIdx.y * 8 + wave) * 16;
  const int KT = K >> 5;

  int arow = rowBase + l16;
  if (arow >= Nrow) arow = Nrow - 1;         // clamp loads; EXEC stays all-ones
  const float* __restrict__ Arow = A + (size_t)arow * K;

  v8f acc[4] = {{}, {}, {}, {}};
  for (int kt = 0; kt < KT; ++kt) {
    const int kk = kt << 5;
    if (kt + 1 < KT)
      __builtin_prefetch(Arow + kk + 32, 0, 3);        // global_prefetch_b8
    // A fragment: two contiguous 8-float runs per lane -> 4 x b128 loads
    const float4 a0 = *(const float4*)(Arow + kk + hi * 8);
    const float4 a1 = *(const float4*)(Arow + kk + hi * 8 + 4);
    const float4 a2 = *(const float4*)(Arow + kk + 16 + hi * 8);
    const float4 a3 = *(const float4*)(Arow + kk + 16 + hi * 8 + 4);
    v16bf av;
    av[0]  = (__bf16)a0.x; av[1]  = (__bf16)a0.y; av[2]  = (__bf16)a0.z; av[3]  = (__bf16)a0.w;
    av[4]  = (__bf16)a1.x; av[5]  = (__bf16)a1.y; av[6]  = (__bf16)a1.z; av[7]  = (__bf16)a1.w;
    av[8]  = (__bf16)a2.x; av[9]  = (__bf16)a2.y; av[10] = (__bf16)a2.z; av[11] = (__bf16)a2.w;
    av[12] = (__bf16)a3.x; av[13] = (__bf16)a3.y; av[14] = (__bf16)a3.z; av[15] = (__bf16)a3.w;
#pragma unroll
    for (int t = 0; t < 4; ++t) {
      const v16bf bv = *(const v16bf*)(Wp +
          ((((size_t)(ct0 + t)) * KT + kt) * 32 + lane) * 16);
      acc[t] = __builtin_amdgcn_wmma_f32_16x16x32_bf16(false, av, false, bv,
                                                       (short)0, acc[t], false, false);
    }
  }
#pragma unroll
  for (int r = 0; r < 8; ++r) {
    const int row = rowBase + r + hi * 8;    // C layout: VGPR r -> M=r / M=r+8
    if (row < Nrow) {
#pragma unroll
      for (int t = 0; t < 4; ++t) {
        const int col = colBase + t * 16 + l16;
        float v = acc[t][r];
        if (bias) v += bias[col];
        if (act == 2) v = v > 0.f ? v : 0.f;
        C[(size_t)row * M + col] = v;
      }
    }
  }
}

// ---------------------------------------------------------------------------
// Elementwise / helper kernels
// ---------------------------------------------------------------------------
__global__ void ema_kernel(const float* __restrict__ pq, const float* __restrict__ pk,
                           float* __restrict__ out, int n) {
  int i = blockIdx.x * blockDim.x + threadIdx.x;
  if (i < n) out[i] = MMT * pk[i] + (1.0f - MMT) * pq[i];
}

__global__ void fill_kernel(float* p, float v, long n) {
  long i = (long)blockIdx.x * blockDim.x + threadIdx.x;
  if (i < n) p[i] = v;
}

__global__ void init_bias_kernel(float* __restrict__ out, const float* __restrict__ bias,
                                 int N, int M) {
  long i = (long)blockIdx.x * blockDim.x + threadIdx.x;
  if (i < (long)N * M) out[i] = bias[i % M];
}

__global__ void elu_kernel(float* p, long n) {
  long i = (long)blockIdx.x * blockDim.x + threadIdx.x;
  if (i < n) { float x = p[i]; p[i] = x > 0.f ? x : (expf(x) - 1.0f); }
}

// a_src[n,h] = sum_c h[n,h,c]*att_src[h,c]; same for a_dst
__global__ void att_reduce_kernel(const float* __restrict__ h,
                                  const float* __restrict__ att_s,
                                  const float* __restrict__ att_d,
                                  float* __restrict__ asrc, float* __restrict__ adst,
                                  int N, int H, int Cc) {
  int i = blockIdx.x * blockDim.x + threadIdx.x;
  if (i >= N * H) return;
  const int hh = i % H;
  const float* hp = h + (size_t)i * Cc;
  const float* as = att_s + hh * Cc;
  const float* ad = att_d + hh * Cc;
  float s1 = 0.f, s2 = 0.f;
  for (int c = 0; c < Cc; ++c) { float v = hp[c]; s1 += v * as[c]; s2 += v * ad[c]; }
  asrc[i] = s1; adst[i] = s2;
}

__device__ __forceinline__ void atomicMaxF(float* addr, float val) {
  int* ai = (int*)addr;
  int cur = __float_as_int(*addr);
  while (val > __int_as_float(cur)) {
    int prev = atomicCAS(ai, cur, __float_as_int(val));
    if (prev == cur) break;
    cur = prev;
  }
}

__device__ __forceinline__ float leaky(float x) {
  return x > 0.f ? x : NEG_SLOPE * x;
}

// segment-max of leaky_relu(a_src[src]+a_dst[dst]) over dst
__global__ void edge_max_kernel(const int* __restrict__ ei, int E, int N, int H,
                                const float* __restrict__ asrc,
                                const float* __restrict__ adst,
                                float* __restrict__ m) {
  long i = (long)blockIdx.x * blockDim.x + threadIdx.x;
  long tot = (long)(E + N) * H;
  if (i >= tot) return;
  long e = i / H; int hh = (int)(i % H);
  int s = (e < E) ? ei[e]     : (int)(e - E);   // self loops appended
  int d = (e < E) ? ei[E + e] : (int)(e - E);
  float x = leaky(asrc[(size_t)s * H + hh] + adst[(size_t)d * H + hh]);
  atomicMaxF(&m[(size_t)d * H + hh], x);
}

// ee = exp(e - m[dst]); z[dst] += ee; stash ee per edge
__global__ void edge_sum_kernel(const int* __restrict__ ei, int E, int N, int H,
                                const float* __restrict__ asrc,
                                const float* __restrict__ adst,
                                const float* __restrict__ m,
                                float* __restrict__ z, float* __restrict__ eebuf) {
  long i = (long)blockIdx.x * blockDim.x + threadIdx.x;
  long tot = (long)(E + N) * H;
  if (i >= tot) return;
  long e = i / H; int hh = (int)(i % H);
  int s = (e < E) ? ei[e]     : (int)(e - E);
  int d = (e < E) ? ei[E + e] : (int)(e - E);
  float x = leaky(asrc[(size_t)s * H + hh] + adst[(size_t)d * H + hh]);
  float ee = expf(x - m[(size_t)d * H + hh]);
  eebuf[i] = ee;
  atomicAdd(&z[(size_t)d * H + hh], ee);
}

// out[dst,h,c] += h[src,h,c] * ee/(z[dst,h]+eps)
__global__ void edge_agg_kernel(const int* __restrict__ ei, int E, int N, int H, int Cc,
                                const float* __restrict__ hbuf,
                                const float* __restrict__ eebuf,
                                const float* __restrict__ z,
                                float* __restrict__ out) {
  long i = (long)blockIdx.x * blockDim.x + threadIdx.x;
  long tot = (long)(E + N) * H * Cc;
  if (i >= tot) return;
  int c = (int)(i % Cc);
  long eh = i / Cc;
  int hh = (int)(eh % H);
  long e = eh / H;
  int s = (e < E) ? ei[e]     : (int)(e - E);
  int d = (e < E) ? ei[E + e] : (int)(e - E);
  float alpha = eebuf[eh] / (z[(size_t)d * H + hh] + 1e-16f);
  atomicAdd(&out[((size_t)d * H + hh) * Cc + c],
            hbuf[((size_t)s * H + hh) * Cc + c] * alpha);
}

__global__ void gather_rows_kernel(const float* __restrict__ src, const int* __restrict__ idx,
                                   float* __restrict__ dst, int B, int C) {
  int i = blockIdx.x * blockDim.x + threadIdx.x;
  if (i >= B * C) return;
  int b = i / C, c = i % C;
  dst[i] = src[(size_t)idx[b] * C + c];
}

__global__ void normalize_rows_kernel(const float* __restrict__ in, float* __restrict__ out,
                                      int B, int C) {
  int b = blockIdx.x * blockDim.x + threadIdx.x;
  if (b >= B) return;
  float s = 0.f;
  for (int c = 0; c < C; ++c) { float v = in[(size_t)b * C + c]; s += v * v; }
  float nrm = sqrtf(s); if (nrm < 1e-12f) nrm = 1e-12f;
  float inv = 1.0f / nrm;
  for (int c = 0; c < C; ++c) out[(size_t)b * C + c] = in[(size_t)b * C + c] * inv;
}

// l_pos + labels (local zeros, global = q_ids)
__global__ void lpos_labels_kernel(const float* __restrict__ q, const float* __restrict__ k,
                                   const int* __restrict__ q_ids,
                                   float* __restrict__ out, long off_lab_l, long off_lab_g) {
  int b = blockIdx.x * blockDim.x + threadIdx.x;
  if (b >= BATCH) return;
  float s = 0.f;
  for (int c = 0; c < NOUT; ++c) s += q[(size_t)b * NOUT + c] * k[(size_t)b * NOUT + c];
  out[(size_t)b * (QSIZE + 1)] = s * INV_TEMP;
  out[off_lab_l + b] = 0.0f;
  out[off_lab_g + b] = (float)q_ids[b];
}

__global__ void logits_local_kernel(const float* __restrict__ q,
                                    const float* __restrict__ queue,
                                    const int* __restrict__ queue_ids,
                                    const int* __restrict__ q_ids,
                                    const int* __restrict__ elem_ids,
                                    float* __restrict__ out) {
  long i = (long)blockIdx.x * blockDim.x + threadIdx.x;
  if (i >= (long)BATCH * QSIZE) return;
  int b = (int)(i / QSIZE), j = (int)(i % QSIZE);
  int qid = q_ids[b];
  const float* qp = q + (size_t)b * NOUT;
  const float* np = queue + ((size_t)qid * NOUT) * QSIZE + j;
  float s = 0.f;
  for (int c = 0; c < NOUT; ++c) s += qp[c] * np[(size_t)c * QSIZE];
  if (queue_ids[(size_t)qid * QSIZE + j] == elem_ids[b]) s = -9e15f;
  out[(size_t)b * (QSIZE + 1) + 1 + j] = s * INV_TEMP;
}

__global__ void queue_mean_kernel(const float* __restrict__ queue, float* __restrict__ qmean) {
  int i = blockIdx.x * blockDim.x + threadIdx.x;   // NQUEUE*NOUT
  if (i >= NQUEUE * NOUT) return;
  const float* p = queue + (size_t)i * QSIZE;
  float s = 0.f;
  for (int j = 0; j < QSIZE; ++j) s += p[j];
  qmean[i] = s * (1.0f / QSIZE);
}

__global__ void logits_global_kernel(const float* __restrict__ q,
                                     const float* __restrict__ qmean,
                                     float* __restrict__ out, long off) {
  int i = blockIdx.x * blockDim.x + threadIdx.x;   // BATCH*NQUEUE
  if (i >= BATCH * NQUEUE) return;
  int b = i / NQUEUE, qi = i % NQUEUE;
  float s = 0.f;
  for (int c = 0; c < NOUT; ++c) s += q[(size_t)b * NOUT + c] * qmean[(size_t)qi * NOUT + c];
  out[off + (size_t)b * NQUEUE + qi] = s * INV_TEMP;
}

// ---------------------------------------------------------------------------
// Host orchestration
// ---------------------------------------------------------------------------
static const int PSIZES[16] = {
  NFEAT * 256, 256, 256, 256,          // W0, att_src0, att_dst0, b0
  256 * 256, 256, 256, 256,            // W1, ...
  256 * NEMB, NEMB, NEMB, NEMB,        // Wout, ...
  NEMB * NEMB, NEMB,                   // projector W1, b1
  NEMB * NOUT, NOUT                    // projector W2, b2
};

struct Scratch {
  float *bufH, *bufO, *asrc, *adst, *mbuf, *zbuf, *eebuf;
  __bf16* wpack;                       // packed weight fragments (<= 256*256 bf16)
};

static void launch_gemm(const float* A, const float* W, const float* bias, float* C,
                        int N, int K, int M, int act, __bf16* Wp, hipStream_t st) {
  const int nth = (M / 16) * (K / 32) * 32;
  pack_w_kernel<<<(nth + 255) / 256, 256, 0, st>>>(W, Wp, K, M);
  gemm_wmma_bf16<<<dim3(M / 64, (N + 127) / 128), 256, 0, st>>>(A, Wp, bias, C, N, K, M, act);
}

static void run_gat_layer(const float* xin, int Kin, const int* ei,
                          const float* Wm, const float* as, const float* ad,
                          const float* bias, int H, int Cc, bool elu,
                          const Scratch& S, hipStream_t st) {
  const int N = NNODE, E = NEDGE, M = H * Cc;
  launch_gemm(xin, Wm, nullptr, S.bufH, N, Kin, M, 0, S.wpack, st);

  int NH = N * H;
  att_reduce_kernel<<<(NH + 255) / 256, 256, 0, st>>>(S.bufH, as, ad, S.asrc, S.adst, N, H, Cc);
  fill_kernel<<<(NH + 255) / 256, 256, 0, st>>>(S.mbuf, -3.0e38f, NH);
  fill_kernel<<<(NH + 255) / 256, 256, 0, st>>>(S.zbuf, 0.0f, NH);

  long EH = (long)(E + N) * H;
  edge_max_kernel<<<(unsigned)((EH + 255) / 256), 256, 0, st>>>(ei, E, N, H, S.asrc, S.adst, S.mbuf);
  edge_sum_kernel<<<(unsigned)((EH + 255) / 256), 256, 0, st>>>(ei, E, N, H, S.asrc, S.adst,
                                                                S.mbuf, S.zbuf, S.eebuf);
  long NM = (long)N * M;
  init_bias_kernel<<<(unsigned)((NM + 255) / 256), 256, 0, st>>>(S.bufO, bias, N, M);
  long tot = (long)(E + N) * H * Cc;
  edge_agg_kernel<<<(unsigned)((tot + 255) / 256), 256, 0, st>>>(ei, E, N, H, Cc,
                                                                 S.bufH, S.eebuf, S.zbuf, S.bufO);
  if (elu) elu_kernel<<<(unsigned)((NM + 255) / 256), 256, 0, st>>>(S.bufO, NM);
}

static void run_branch(const float* x, const int* ei, const int* idx,
                       const float* const P[16], const Scratch& S,
                       float* sel, float* hid, float* out64, float* qn,
                       hipStream_t st) {
  // encoder: 3 GAT layers (concat 4x64, concat 4x64, mean 1x128)
  run_gat_layer(x,      NFEAT, ei, P[0], P[1],  P[2],  P[3],  4, 64,  true,  S, st);
  run_gat_layer(S.bufO, 256,   ei, P[4], P[5],  P[6],  P[7],  4, 64,  true,  S, st);
  run_gat_layer(S.bufO, 256,   ei, P[8], P[9],  P[10], P[11], 1, NEMB, false, S, st);
  // projector + normalize are row-wise and commute with the batch gather:
  // gather the 512 selected rows first -> 39x less projector work.
  gather_rows_kernel<<<(BATCH * NEMB + 255) / 256, 256, 0, st>>>(S.bufO, idx, sel, BATCH, NEMB);
  launch_gemm(sel, P[12], P[13], hid,   BATCH, NEMB, NEMB, 2, S.wpack, st);
  launch_gemm(hid, P[14], P[15], out64, BATCH, NEMB, NOUT, 0, S.wpack, st);
  normalize_rows_kernel<<<(BATCH + 255) / 256, 256, 0, st>>>(out64, qn, BATCH, NOUT);
}

extern "C" void kernel_launch(void* const* d_in, const int* in_sizes, int n_in,
                              void* d_out, int out_size, void* d_ws, size_t ws_size,
                              hipStream_t stream) {
  // ---- inputs ----
  const float* inputs_q = (const float*)d_in[0];
  const int*   ei_q     = (const int*)d_in[1];
  const int*   idx_q    = (const int*)d_in[2];
  const float* inputs_k = (const float*)d_in[3];
  const int*   ei_k     = (const int*)d_in[4];
  const int*   idx_k    = (const int*)d_in[5];
  const int*   q_ids    = (const int*)d_in[6];
  const int*   elem_ids = (const int*)d_in[7];

  const float* Pq[16]; const float* Pk[16];
  const float* queue; const int* queue_ids;
  if (n_in >= 42) {                       // params flattened as separate tensors
    for (int i = 0; i < 16; ++i) Pq[i] = (const float*)d_in[8 + i];
    for (int i = 0; i < 16; ++i) Pk[i] = (const float*)d_in[24 + i];
    queue     = (const float*)d_in[40];
    queue_ids = (const int*)d_in[41];
  } else {                                // params packed contiguously
    const float* bq = (const float*)d_in[8];
    const float* bk = (const float*)d_in[9];
    size_t o = 0;
    for (int i = 0; i < 16; ++i) { Pq[i] = bq + o; Pk[i] = bk + o; o += PSIZES[i]; }
    queue     = (const float*)d_in[10];
    queue_ids = (const int*)d_in[11];
  }

  // ---- workspace layout (floats; every block is a multiple of 32 bytes) ----
  float* w = (float*)d_ws;
  size_t off = 0;
  auto alloc = [&](size_t n) { float* p = w + off; off += n; return p; };

  size_t ptot = 0; for (int i = 0; i < 16; ++i) ptot += PSIZES[i];
  float* pkbuf = alloc(ptot);
  const float* Pe[16];                    // EMA-updated key params
  { size_t o = 0; for (int i = 0; i < 16; ++i) { Pe[i] = pkbuf + o; o += PSIZES[i]; } }

  Scratch S;
  S.bufH  = alloc((size_t)NNODE * 256);
  S.bufO  = alloc((size_t)NNODE * 256);
  S.asrc  = alloc((size_t)NNODE * 4);
  S.adst  = alloc((size_t)NNODE * 4);
  S.mbuf  = alloc((size_t)NNODE * 4);
  S.zbuf  = alloc((size_t)NNODE * 4);
  S.eebuf = alloc((size_t)(NEDGE + NNODE) * 4);
  float* sel   = alloc((size_t)BATCH * NEMB);
  float* hid   = alloc((size_t)BATCH * NEMB);
  float* out64 = alloc((size_t)BATCH * NOUT);
  float* qvec  = alloc((size_t)BATCH * NOUT);
  float* kvec  = alloc((size_t)BATCH * NOUT);
  float* qmean = alloc((size_t)NQUEUE * NOUT);
  S.wpack = (__bf16*)alloc(256 * 256 / 2);   // 65536 bf16 = 128 KB pack buffer
  (void)ws_size; (void)in_sizes; (void)out_size;

  // ---- q branch (params_q) ----
  run_branch(inputs_q, ei_q, idx_q, Pq, S, sel, hid, out64, qvec, stream);

  // ---- EMA update of key params, then k branch ----
  { size_t o = 0;
    for (int i = 0; i < 16; ++i) {
      ema_kernel<<<(PSIZES[i] + 255) / 256, 256, 0, stream>>>(Pq[i], Pk[i], pkbuf + o, PSIZES[i]);
      o += PSIZES[i];
    } }
  run_branch(inputs_k, ei_k, idx_k, Pe, S, sel, hid, out64, kvec, stream);

  // ---- contrastive logits ----
  float* out = (float*)d_out;
  const long OFF_LAB_L = (long)BATCH * (QSIZE + 1);
  const long OFF_LOG_G = OFF_LAB_L + BATCH;
  const long OFF_LAB_G = OFF_LOG_G + (long)BATCH * NQUEUE;

  lpos_labels_kernel<<<(BATCH + 255) / 256, 256, 0, stream>>>(qvec, kvec, q_ids, out,
                                                              OFF_LAB_L, OFF_LAB_G);
  long nloc = (long)BATCH * QSIZE;
  logits_local_kernel<<<(unsigned)((nloc + 255) / 256), 256, 0, stream>>>(
      qvec, queue, queue_ids, q_ids, elem_ids, out);
  queue_mean_kernel<<<(NQUEUE * NOUT + 255) / 256, 256, 0, stream>>>(queue, qmean);
  logits_global_kernel<<<(BATCH * NQUEUE + 255) / 256, 256, 0, stream>>>(qvec, qmean, out,
                                                                         OFF_LOG_G);
}
